// UniversalVocoder_52604759441888
// MI455X (gfx1250) — matrix operand order repaired and checked by reading, not compiled, not run on hardware
//
#include <hip/hip_runtime.h>
#include <hip/hip_bf16.h>

#define TM 32
#define BB 4
#define DM 80
#define DE 128
#define DV 256
#define DD 896
#define DP 512
#define NB 256                 // N_BINS
#define HOP 80
#define TW (TM*HOP)            // 2560
#define ROWS (TW*BB)           // 10240
#define G3 (3*DD)              // 2688
#define KIT_DD (DD/32)         // 28
#define HT_TILES (DD/16)       // 56
#define NT_G3 (G3/16)          // 168

typedef __attribute__((ext_vector_type(16))) __bf16 bf16x16;
typedef __attribute__((ext_vector_type(8)))  float  v8f;

union BfPack { uint4 u[2]; bf16x16 v; };

static __device__ __forceinline__ v8f wmma_bf16(bf16x16 a, bf16x16 b, v8f c) {
  // (neg_a, A, neg_b, B, c_mod, C, reuse_a, reuse_b)
  return __builtin_amdgcn_wmma_f32_16x16x32_bf16(false, a, false, b, (short)0, c, false, false);
}

// A operand 16x32 bf16 (row-major src, rows=M, ld elems).
// Order used here (same as B): lanes 0-15 hold K=0..15; lanes 16-31 hold K=16..31 (contiguous).
static __device__ __forceinline__ bf16x16 ld_a_bf16(const __bf16* base, int ld) {
  int lane = threadIdx.x & 31;
  const __bf16* p = base + (size_t)(lane & 15) * ld + ((lane >> 4) * 16);
  BfPack t;
  t.u[0] = *reinterpret_cast<const uint4*>(p);
  t.u[1] = *reinterpret_cast<const uint4*>(p + 8);
  return t.v;
}
static __device__ __forceinline__ bf16x16 ld_a_f32(const float* base, int ld) {
  int lane = threadIdx.x & 31;
  const float* p = base + (size_t)(lane & 15) * ld + ((lane >> 4) * 16);
  bf16x16 r;
#pragma unroll
  for (int i = 0; i < 8; ++i) r[i] = (__bf16)p[i];
#pragma unroll
  for (int i = 0; i < 8; ++i) r[i + 8] = (__bf16)p[i + 8];
  return r;
}
// B operand 32x16 bf16 from W (N rows x K cols, row-major): lane n = column n of B
// = row n of W; lanes 0-15 carry K=0..15, lanes 16-31 carry K=16..31 (contiguous).
static __device__ __forceinline__ bf16x16 ld_b_bf16(const __bf16* base, int ld) {
  int lane = threadIdx.x & 31;
  const __bf16* p = base + (size_t)(lane & 15) * ld + ((lane >> 4) * 16);
  BfPack t;
  t.u[0] = reinterpret_cast<const uint4*>(p)[0];
  t.u[1] = reinterpret_cast<const uint4*>(p)[1];
  return t.v;
}
static __device__ __forceinline__ bf16x16 ld_b_f32(const float* base, int ld) {
  int lane = threadIdx.x & 31;
  const float* p = base + (size_t)(lane & 15) * ld + ((lane >> 4) * 16);
  bf16x16 r;
#pragma unroll
  for (int i = 0; i < 16; ++i) r[i] = (__bf16)p[i];
  return r;
}

static __device__ __forceinline__ float sigf(float x) { return 1.0f / (1.0f + __expf(-x)); }
static __device__ __forceinline__ int quant(float v) {
  float f = floorf((v + 1.0f) * (NB * 0.5f));
  f = fminf(fmaxf(f, 0.0f), (float)(NB - 1));
  return (int)f;
}

// ---------------- small utility kernels ----------------
__global__ void k_f2bf(const float* __restrict__ s, __bf16* __restrict__ d, int n) {
  int i = blockIdx.x * blockDim.x + threadIdx.x;
  if (i < n) d[i] = (__bf16)s[i];
}
__global__ void k_quant(const float* __restrict__ y, int* __restrict__ q, int* __restrict__ qt) {
  int i = blockIdx.x * blockDim.x + threadIdx.x;
  if (i >= ROWS) return;
  float wv = (i < BB) ? 0.0f : y[i - BB];    // teacher-forcing right shift
  q[i]  = quant(wv);
  qt[i] = quant(y[i]);
}

// ---------------- encoder GRU (one workgroup per direction) ----------------
__global__ __launch_bounds__(256)
void k_enc_gru(const float* __restrict__ xin, int I,
               const float* fWih, const float* fWhh, const float* fbih, const float* fbhh,
               const float* bWih, const float* bWhh, const float* bbih, const float* bbhh,
               float* __restrict__ out) {
  const int dir = blockIdx.x;
  const float* Wih = dir ? bWih : fWih;
  const float* Whh = dir ? bWhh : fWhh;
  const float* bih = dir ? bbih : fbih;
  const float* bhh = dir ? bbhh : fbhh;
  __shared__ float h[BB * DE];
  __shared__ float ig[BB * 3 * DE];
  __shared__ float hg[BB * 3 * DE];
  const int tid = threadIdx.x;
  for (int i = tid; i < BB * DE; i += 256) h[i] = 0.0f;
  __syncthreads();
  for (int s = 0; s < TM; ++s) {
    const int t = dir ? (TM - 1 - s) : s;
    for (int idx = tid; idx < BB * 3 * DE; idx += 256) {
      int m = idx / (3 * DE), n = idx % (3 * DE);
      const float* xr = xin + (size_t)t * (BB * I) + m * I;
      const float* wr = Wih + (size_t)n * I;
      float si = bih[n];
      for (int k = 0; k < I; ++k) si += xr[k] * wr[k];
      ig[idx] = si;
      const float* hr = h + m * DE;
      const float* ur = Whh + (size_t)n * DE;
      float sh = bhh[n];
      for (int k = 0; k < DE; ++k) sh += hr[k] * ur[k];
      hg[idx] = sh;
    }
    __syncthreads();
    for (int idx = tid; idx < BB * DE; idx += 256) {
      int m = idx / DE, j = idx % DE;
      float r = sigf(ig[m * 3 * DE + j] + hg[m * 3 * DE + j]);
      float z = sigf(ig[m * 3 * DE + DE + j] + hg[m * 3 * DE + DE + j]);
      float n = tanhf(ig[m * 3 * DE + 2 * DE + j] + r * hg[m * 3 * DE + 2 * DE + j]);
      float hn = (1.0f - z) * n + z * h[idx];
      h[idx] = hn;
      out[(size_t)t * (BB * 2 * DE) + m * (2 * DE) + dir * DE + j] = hn;
    }
    __syncthreads();
  }
}

// ---------------- generic 16x16-tile WMMA GEMM: C = A(f32) @ W(f32)^T + bias ----------------
__global__ void k_gemm16(const float* __restrict__ A, int lda,
                         const float* __restrict__ W, int ldw,
                         const float* __restrict__ bias,
                         float* __restrict__ C, int ldc, int ntiles, int kiters) {
  const int mt = blockIdx.x / ntiles, nt = blockIdx.x % ntiles;
  v8f acc = {};
  for (int k = 0; k < kiters; ++k) {
    bf16x16 a = ld_a_f32(A + (size_t)mt * 16 * lda + k * 32, lda);
    bf16x16 b = ld_b_f32(W + (size_t)nt * 16 * ldw + k * 32, ldw);
    acc = wmma_bf16(a, b, acc);
  }
  const int lane = threadIdx.x & 31;
  const int col = lane & 15, rb = (lane >> 4) << 3;
#pragma unroll
  for (int v = 0; v < 8; ++v) {
    int m = mt * 16 + v + rb, n = nt * 16 + col;
    float bv = bias ? bias[n] : 0.0f;
    C[(size_t)m * ldc + n] = acc[v] + bv;
  }
}

// ---------------- persistent wav-GRU: 1 WGP, 32 waves, h in LDS ----------------
__global__ __launch_bounds__(1024)
void k_wav_gru(const float* __restrict__ EW, const float* __restrict__ CW,
               const __bf16* __restrict__ Whh, const float* __restrict__ bhh,
               const int* __restrict__ q, __bf16* __restrict__ o) {
  __shared__ float hf[16 * DD];                 // 56 KB, rows 4..15 stay zero
  const int tid = threadIdx.x;
  for (int i = tid; i < 16 * DD; i += 1024) hf[i] = 0.0f;
  __syncthreads();
  const int wave = tid >> 5, lane = tid & 31, col = lane & 15;
  for (int t = 0; t < TW; ++t) {
    float hnew[2][4];
    const int crow = (t / HOP) * BB;
#pragma unroll
    for (int tix = 0; tix < 2; ++tix) {
      const int ht = wave + 32 * tix;
      if (ht < HT_TILES) {
        v8f ar = {}, az = {}, an = {};
        const __bf16* wr = Whh + (size_t)(0 * DD + ht * 16) * DD;
        const __bf16* wz = Whh + (size_t)(1 * DD + ht * 16) * DD;
        const __bf16* wn = Whh + (size_t)(2 * DD + ht * 16) * DD;
        for (int k = 0; k < KIT_DD; ++k) {
          bf16x16 a = ld_a_f32(hf + k * 32, DD);
          ar = wmma_bf16(a, ld_b_bf16(wr + k * 32, DD), ar);
          az = wmma_bf16(a, ld_b_bf16(wz + k * 32, DD), az);
          an = wmma_bf16(a, ld_b_bf16(wn + k * 32, DD), an);
          __builtin_prefetch(wr + (k + 1) * 32, 0, 1);   // global_prefetch_b8
        }
        if (lane < 16) {                       // C rows 0..7 live in lanes 0..15
          const int j = ht * 16 + col;
          const float bhr = bhh[j], bhz = bhh[DD + j], bhn = bhh[2 * DD + j];
#pragma unroll
          for (int v = 0; v < 4; ++v) {        // batch rows m = v (0..3)
            const int qi = q[t * BB + v];
            const float* ew = EW + (size_t)qi * G3;
            const float* cw = CW + (size_t)(crow + v) * G3;   // bih folded into CW
            float r = sigf(ew[j]          + cw[j]          + ar[v] + bhr);
            float z = sigf(ew[DD + j]     + cw[DD + j]     + az[v] + bhz);
            float n = tanhf(ew[2 * DD + j] + cw[2 * DD + j] + r * (an[v] + bhn));
            hnew[tix][v] = (1.0f - z) * n + z * hf[v * DD + j];
          }
        }
      }
    }
    __syncthreads();                            // all reads of old h complete
#pragma unroll
    for (int tix = 0; tix < 2; ++tix) {
      const int ht = wave + 32 * tix;
      if (ht < HT_TILES && lane < 16) {
        const int j = ht * 16 + col;
#pragma unroll
        for (int v = 0; v < 4; ++v) {
          hf[v * DD + j] = hnew[tix][v];
          o[(size_t)(t * BB + v) * DD + j] = (__bf16)hnew[tix][v];
        }
      }
    }
    __syncthreads();
  }
}

// ---------------- fused projection + GELU + logits + log-softmax NLL ----------------
__global__ __launch_bounds__(256)
void k_head(const __bf16* __restrict__ o, const float* __restrict__ Wp,
            const float* __restrict__ bp, const float* __restrict__ Wd,
            const float* __restrict__ bd, const int* __restrict__ qt,
            float* __restrict__ partial) {
  __shared__ __align__(16) __bf16 pbf[16 * DP];   // 16 KB
  __shared__ float lg[16 * NB];                   // 16 KB
  __shared__ float red[256];
  const int tid = threadIdx.x, wave = tid >> 5, lane = tid & 31;
  const int col = lane & 15, rb = (lane >> 4) << 3;
  const int r0 = blockIdx.x * 16;
  // projection tile (16 x 512) with GELU -> bf16 LDS
  for (int nt = wave; nt < DP / 16; nt += 8) {
    v8f acc = {};
    for (int k = 0; k < KIT_DD; ++k) {
      bf16x16 a = ld_a_bf16(o + (size_t)r0 * DD + k * 32, DD);
      bf16x16 b = ld_b_f32(Wp + (size_t)(nt * 16) * DD + k * 32, DD);
      acc = wmma_bf16(a, b, acc);
    }
#pragma unroll
    for (int v = 0; v < 8; ++v) {
      int m = v + rb, n = nt * 16 + col;
      float x = acc[v] + bp[n];
      float g = 0.5f * x * (1.0f + erff(x * 0.70710678118f));   // exact GELU
      pbf[m * DP + n] = (__bf16)g;
    }
  }
  __syncthreads();
  // logits tile (16 x 256)
  for (int nt = wave; nt < NB / 16; nt += 8) {
    v8f acc = {};
    for (int k = 0; k < DP / 32; ++k) {
      bf16x16 a = ld_a_bf16((const __bf16*)pbf + k * 32, DP);
      bf16x16 b = ld_b_f32(Wd + (size_t)(nt * 16) * DP + k * 32, DP);
      acc = wmma_bf16(a, b, acc);
    }
#pragma unroll
    for (int v = 0; v < 8; ++v)
      lg[(v + rb) * NB + nt * 16 + col] = acc[v] + bd[nt * 16 + col];
  }
  __syncthreads();
  // per-row log-softmax + target gather
  float bsum = 0.0f;
  for (int r = 0; r < 16; ++r) {
    float val = lg[r * NB + tid];
    red[tid] = val; __syncthreads();
    for (int s = 128; s > 0; s >>= 1) { if (tid < s) red[tid] = fmaxf(red[tid], red[tid + s]); __syncthreads(); }
    float mx = red[0]; __syncthreads();
    red[tid] = __expf(val - mx); __syncthreads();
    for (int s = 128; s > 0; s >>= 1) { if (tid < s) red[tid] += red[tid + s]; __syncthreads(); }
    if (tid == 0) {
      int tg = qt[r0 + r];
      bsum += mx + __logf(red[0]) - lg[r * NB + tg];
    }
    __syncthreads();
  }
  if (tid == 0) partial[blockIdx.x] = bsum;
}

__global__ void k_reduce(const float* __restrict__ partial, float* __restrict__ out) {
  __shared__ float red[256];
  const int tid = threadIdx.x;
  float s = 0.0f;
  for (int i = tid; i < ROWS / 16; i += 256) s += partial[i];
  red[tid] = s; __syncthreads();
  for (int k = 128; k > 0; k >>= 1) { if (tid < k) red[tid] += red[tid + k]; __syncthreads(); }
  if (tid == 0) out[0] = red[0] / (float)ROWS;
}

extern "C" void kernel_launch(void* const* d_in, const int* in_sizes, int n_in,
                              void* d_out, int out_size, void* d_ws, size_t ws_size,
                              hipStream_t stream) {
  (void)in_sizes; (void)n_in; (void)out_size; (void)ws_size;
  auto F = [&](int i) { return (const float*)d_in[i]; };
  const float* x    = F(0);
  const float* y    = F(1);
  const float* wWih = F(18), *wWhh = F(19), *wbih = F(20), *wbhh = F(21);
  const float* Wp = F(22), *bp = F(23), *E = F(24), *Wd = F(25), *bd = F(26);

  char* cur = (char*)d_ws;
  auto alloc = [&](size_t bytes) { void* p = (void*)cur; cur += (bytes + 255) & ~(size_t)255; return p; };
  float*  c0      = (float*) alloc((size_t)TM * BB * 2 * DE * 4);
  float*  c1      = (float*) alloc((size_t)TM * BB * 2 * DE * 4);
  float*  EW      = (float*) alloc((size_t)NB * G3 * 4);
  float*  CW      = (float*) alloc((size_t)TM * BB * G3 * 4);
  __bf16* WhhB    = (__bf16*)alloc((size_t)G3 * DD * 2);
  int*    q       = (int*)   alloc((size_t)ROWS * 4);
  int*    qt      = (int*)   alloc((size_t)ROWS * 4);
  __bf16* ob      = (__bf16*)alloc((size_t)ROWS * DD * 2);
  float*  partial = (float*) alloc((size_t)(ROWS / 16) * 4);

  // encoder: 2 bidirectional layers
  k_enc_gru<<<2, 256, 0, stream>>>(x, DM, F(2), F(3), F(4), F(5), F(6), F(7), F(8), F(9), c0);
  k_enc_gru<<<2, 256, 0, stream>>>(c0, 2 * DE, F(10), F(11), F(12), F(13), F(14), F(15), F(16), F(17), c1);
  // weight prep + quantization
  k_f2bf<<<(G3 * DD + 255) / 256, 256, 0, stream>>>(wWhh, WhhB, G3 * DD);
  k_quant<<<(ROWS + 255) / 256, 256, 0, stream>>>(y, q, qt);
  // folded input projections: EW = E @ Wih[:, :256]^T ; CW = c1 @ Wih[:, 256:]^T + bih
  k_gemm16<<<(NB / 16) * NT_G3, 32, 0, stream>>>(E, DV, wWih, DV + 2 * DE, nullptr, EW, G3, NT_G3, DV / 32);
  k_gemm16<<<(TM * BB / 16) * NT_G3, 32, 0, stream>>>(c1, 2 * DE, wWih + DV, DV + 2 * DE, wbih, CW, G3, NT_G3, (2 * DE) / 32);
  // sequential wav GRU (persistent, WMMA recurrent matmul)
  k_wav_gru<<<1, 1024, 0, stream>>>(EW, CW, WhhB, wbhh, q, ob);
  // fused head + deterministic reduction
  k_head<<<ROWS / 16, 256, 0, stream>>>(ob, Wp, bp, Wd, bd, qt, partial);
  k_reduce<<<1, 256, 0, stream>>>(partial, (float*)d_out);
}
